// FastWeightCDEv2_84859963834878
// MI455X (gfx1250) — matrix operand order repaired and checked by reading, not compiled, not run on hardware
//
#include <hip/hip_runtime.h>
#include <hip/hip_bf16.h>

// MI455X (gfx1250) implementation of FastWeightCDEv2.
// Phase 1: batched bf16 WMMA GEMMs for all 17 kv evaluations (W-independent).
// Phase 2: per-(b,h) serial RK4 Oja scan, W column-per-lane in VGPRs (wave32).
// Phase 3: readout GEMMs (out_proj / ff1 / ff2 / final) with WMMA.

typedef __attribute__((ext_vector_type(16))) __bf16 v16bf;
typedef __attribute__((ext_vector_type(8)))  float  v8f;

__device__ __forceinline__ __bf16 f2bf(float x) { return (__bf16)x; }

// ---------------------------------------------------------------------------
// Generic GEMM: C[M,N] = A[M,K] @ W[N,K]^T + bias[N] (+residual) (relu?)
// Block = 256 threads = 8 waves arranged 2(M) x 4(N).
// Wave tile = 16(M) x 32(N): two v_wmma accumulators share one A fragment.
// Block tile = 32 x 128. REQUIRES M % 32 == 0, N % 128 == 0, K % 32 == 0
// (true for every call in this model) -> no store guards, straight-line
// epilogue, relu/residual resolved at compile time.
// ---------------------------------------------------------------------------
template <int RELU, int RES>
__global__ __launch_bounds__(256)
void gemm_bf16(const float* __restrict__ A, int lda,
               const float* __restrict__ W,        // [N, K] row-major (fout, fin)
               const float* __restrict__ bias,     // [N]
               const float* __restrict__ residual, // [M, ldc] or nullptr
               float* __restrict__ C, int ldc, int K)
{
  const int lane  = threadIdx.x & 31;
  const int warp  = threadIdx.x >> 5;
  const int warpM = warp >> 2;
  const int warpN = warp & 3;
  const int m0 = blockIdx.y * 32 + warpM * 16;
  const int n0 = blockIdx.x * 128 + warpN * 32;
  const int lr = lane & 15;        // A row within tile / B col within tile
  const int hi = lane >> 4;
  const int klo   = hi ? 8  : 0;   // unused: A is loaded with kbase, same as B
  const int kbase = hi ? 16 : 0;   // A- and B-fragment K sub-offset (contiguous k = 16*hi + j)

  v8f acc0 = {}, acc1 = {};
  const float* Arow  = A + (size_t)(m0 + lr) * lda;
  const float* Brow0 = W + (size_t)(n0 + lr) * K;
  const float* Brow1 = W + (size_t)(n0 + 16 + lr) * K;

  for (int kk = 0; kk < K; kk += 32) {
    // speculative prefetch of next K-slab (global_prefetch_b8; OOB is dropped)
    __builtin_prefetch(Arow  + kk + 64, 0, 1);
    __builtin_prefetch(Brow0 + kk + 64, 0, 1);
    __builtin_prefetch(Brow1 + kk + 64, 0, 1);
    v16bf a, b0, b1;
#pragma unroll
    for (int j = 0; j < 16; ++j) a[j]     = f2bf(Arow[kk + kbase + j]);
#pragma unroll
    for (int j = 0; j < 16; ++j) b0[j]    = f2bf(Brow0[kk + kbase + j]);
#pragma unroll
    for (int j = 0; j < 16; ++j) b1[j]    = f2bf(Brow1[kk + kbase + j]);
    acc0 = __builtin_amdgcn_wmma_f32_16x16x32_bf16(false, a, false, b0,
                                                   (short)0, acc0, false, false);
    acc1 = __builtin_amdgcn_wmma_f32_16x16x32_bf16(false, a, false, b1,
                                                   (short)0, acc1, false, false);
  }

  // C/D layout: lane -> n = lr (per 16-wide tile); VGPR r -> m = m0 + 8*hi + r
  const int nA = n0 + lr;
  const int nB = n0 + 16 + lr;
  const float biasA = bias[nA];
  const float biasB = bias[nB];
  const int mb = m0 + hi * 8;
#pragma unroll
  for (int r = 0; r < 8; ++r) {
    const size_t row = (size_t)(mb + r) * ldc;
    float v0 = acc0[r] + biasA;
    float v1 = acc1[r] + biasB;
    if (RES) { v0 += residual[row + nA]; v1 += residual[row + nB]; }
    if (RELU) { v0 = fmaxf(v0, 0.f); v1 = fmaxf(v1, 0.f); }
    C[row + nA] = v0;
    C[row + nB] = v1;
  }
}

// ---------------------------------------------------------------------------
// Pack logsig (B,T,128) into DX rows 32..543  (row = (1+t)*32 + b)
// ---------------------------------------------------------------------------
__global__ __launch_bounds__(256)
void pack_logsig(const float* __restrict__ logsig, float* __restrict__ DX)
{
  int gid = blockIdx.x * 256 + threadIdx.x;
  if (gid >= 32 * 16 * 128) return;
  int c = gid & 127;
  int r = gid >> 7;        // b*16 + t
  int t = r & 15;
  int b = r >> 4;
  DX[((size_t)((1 + t) * 32 + b)) * 128 + c] = logsig[gid];
}

// ---------------------------------------------------------------------------
// Row LayerNorm over 1024 columns. 256 threads, 4 cols/thread.
// ---------------------------------------------------------------------------
__global__ __launch_bounds__(256)
void ln_rows_1024(const float* __restrict__ src, float* __restrict__ dst,
                  const float* __restrict__ g, const float* __restrict__ bb)
{
  __shared__ float red[256];
  const int row = blockIdx.x;
  const float* x = src + (size_t)row * 1024;
  float v4[4];
  float s = 0.f;
#pragma unroll
  for (int i = 0; i < 4; ++i) { v4[i] = x[threadIdx.x + 256 * i]; s += v4[i]; }
  red[threadIdx.x] = s; __syncthreads();
  for (int o = 128; o > 0; o >>= 1) {
    if (threadIdx.x < o) red[threadIdx.x] += red[threadIdx.x + o];
    __syncthreads();
  }
  const float mean = red[0] * (1.f / 1024.f);
  __syncthreads();
  s = 0.f;
#pragma unroll
  for (int i = 0; i < 4; ++i) { float d = v4[i] - mean; s += d * d; }
  red[threadIdx.x] = s; __syncthreads();
  for (int o = 128; o > 0; o >>= 1) {
    if (threadIdx.x < o) red[threadIdx.x] += red[threadIdx.x + o];
    __syncthreads();
  }
  const float rstd = rsqrtf(red[0] * (1.f / 1024.f) + 1e-5f);
#pragma unroll
  for (int i = 0; i < 4; ++i) {
    int c = threadIdx.x + 256 * i;
    dst[(size_t)row * 1024 + c] = (v4[i] - mean) * rstd * g[c] + bb[c];
  }
}

// ---------------------------------------------------------------------------
// In-place softmax over contiguous 64-float head segments. 64 threads.
// ---------------------------------------------------------------------------
__global__ __launch_bounds__(64)
void softmax_head64(float* __restrict__ p)
{
  __shared__ float red[64];
  float* x = p + (size_t)blockIdx.x * 64;
  float v = x[threadIdx.x];
  red[threadIdx.x] = v; __syncthreads();
  for (int o = 32; o > 0; o >>= 1) {
    if (threadIdx.x < o) red[threadIdx.x] = fmaxf(red[threadIdx.x], red[threadIdx.x + o]);
    __syncthreads();
  }
  const float mx = red[0];
  __syncthreads();
  const float e = __expf(v - mx);
  red[threadIdx.x] = e; __syncthreads();
  for (int o = 32; o > 0; o >>= 1) {
    if (threadIdx.x < o) red[threadIdx.x] += red[threadIdx.x + o];
    __syncthreads();
  }
  x[threadIdx.x] = e / red[0];
}

// ---------------------------------------------------------------------------
// In-place LayerNorm over 64-float head segments with D-sized gamma/beta.
// ---------------------------------------------------------------------------
__global__ __launch_bounds__(64)
void ln_head64(float* __restrict__ p, const float* __restrict__ g,
               const float* __restrict__ bb)
{
  __shared__ float red[64];
  float* x = p + (size_t)blockIdx.x * 64;
  const float v = x[threadIdx.x];
  red[threadIdx.x] = v; __syncthreads();
  for (int o = 32; o > 0; o >>= 1) {
    if (threadIdx.x < o) red[threadIdx.x] += red[threadIdx.x + o];
    __syncthreads();
  }
  const float mean = red[0] * (1.f / 64.f);
  __syncthreads();
  const float d = v - mean;
  red[threadIdx.x] = d * d; __syncthreads();
  for (int o = 32; o > 0; o >>= 1) {
    if (threadIdx.x < o) red[threadIdx.x] += red[threadIdx.x + o];
    __syncthreads();
  }
  const float rstd = rsqrtf(red[0] * (1.f / 64.f) + 1e-5f);
  x[threadIdx.x] = d * rstd * g[threadIdx.x] + bb[threadIdx.x];
}

// ---------------------------------------------------------------------------
// RK4 Oja scan. One block (64 threads = 2 wave32) per (b,h).
// Thread j owns column j of W (64 fp32 in VGPRs). All substep fields are
// rank-1 with left vector v, so column j of each field is v * u_j.
// Ends with z_i = sum_j W[i][j] * q_j via an LDS transpose (stride 65,
// bank-conflict-free both directions).
// ---------------------------------------------------------------------------
__global__ __launch_bounds__(64)
void oja_scan(const float* __restrict__ Kb, const float* __restrict__ Vb,
              const float* __restrict__ Qb, float* __restrict__ z)
{
  __shared__ float vs[64], ks[64], vps[64], kps[64], qs[64];
  __shared__ float Wsh[64 * 65];
  const int b = blockIdx.x >> 4;
  const int h = blockIdx.x & 15;
  const int j = threadIdx.x;
  const size_t hoff = (size_t)h * 64;

  float Wc[64];

  // W0 = v0 (x) k0   (slice 0 = initial projection)
  {
    const size_t r0 = (size_t)b * 1024 + hoff;
    vs[j] = Vb[r0 + j];
    ks[j] = Kb[r0 + j];
    __syncthreads();
    const float kj = ks[j];
#pragma unroll
    for (int i = 0; i < 64; ++i) Wc[i] = vs[i] * kj;
  }

  for (int t = 0; t < 16; ++t) {
    const int s0 = 1 + t;
    const int s1 = (t + 2 <= 16) ? (t + 2) : 16;   // 1 + min(t+1, 15)
    __syncthreads();
    vs[j]  = Vb[((size_t)(s0 * 32 + b)) * 1024 + hoff + j];
    ks[j]  = Kb[((size_t)(s0 * 32 + b)) * 1024 + hoff + j];
    vps[j] = Vb[((size_t)(s1 * 32 + b)) * 1024 + hoff + j];
    kps[j] = Kb[((size_t)(s1 * 32 + b)) * 1024 + hoff + j];
    __syncthreads();

    float a1 = 0.f;
#pragma unroll
    for (int i = 0; i < 64; ++i) a1 = fmaf(vs[i], Wc[i], a1);
    const float u1 = ks[j] - a1;

    float a2 = 0.f;
#pragma unroll
    for (int i = 0; i < 64; ++i) a2 = fmaf(vs[i], fmaf(0.5f * u1, vs[i], Wc[i]), a2);
    const float u2 = ks[j] - a2;

    float a3 = 0.f;
#pragma unroll
    for (int i = 0; i < 64; ++i) a3 = fmaf(vs[i], fmaf(0.5f * u2, vs[i], Wc[i]), a3);
    const float u3 = ks[j] - a3;

    float a4 = 0.f;
#pragma unroll
    for (int i = 0; i < 64; ++i) a4 = fmaf(vps[i], fmaf(u3, vs[i], Wc[i]), a4);
    const float u4 = kps[j] - a4;

    const float c1 = (u1 + 2.f * (u2 + u3)) * (1.f / 6.f);
    const float c4 = u4 * (1.f / 6.f);
#pragma unroll
    for (int i = 0; i < 64; ++i) Wc[i] = fmaf(vps[i], c4, fmaf(vs[i], c1, Wc[i]));
  }

  // Readout: z_i = sum_j W[i][j] * q_j
  __syncthreads();
  qs[j] = Qb[(size_t)b * 1024 + hoff + j];
#pragma unroll
  for (int i = 0; i < 64; ++i) Wsh[j * 65 + i] = Wc[i];
  __syncthreads();
  float zi = 0.f;
#pragma unroll
  for (int jj = 0; jj < 64; ++jj) zi = fmaf(Wsh[jj * 65 + j], qs[jj], zi);
  z[(size_t)b * 1024 + hoff + j] = zi;
}

// ---------------------------------------------------------------------------
extern "C" void kernel_launch(void* const* d_in, const int* in_sizes, int n_in,
                              void* d_out, int out_size, void* d_ws, size_t ws_size,
                              hipStream_t stream)
{
  (void)in_sizes; (void)n_in; (void)out_size; (void)ws_size;
  // setup_inputs() flat order (depth-first dict order):
  const float* initial = (const float*)d_in[0];   // 32x64
  const float* logsig  = (const float*)d_in[1];   // 32x16x128
  const float* w_init  = (const float*)d_in[2];   // 128x64
  const float* b_init  = (const float*)d_in[3];
  const float* w_in    = (const float*)d_in[4];   // 1024x128
  const float* b_in    = (const float*)d_in[5];
  const float* g_inln  = (const float*)d_in[6];
  const float* b_inln  = (const float*)d_in[7];
  const float* w_k     = (const float*)d_in[8];   // 1024x1024
  const float* b_k     = (const float*)d_in[9];
  const float* w_v     = (const float*)d_in[10];  // 1024x1024
  const float* b_v     = (const float*)d_in[11];
  const float* g_vln   = (const float*)d_in[12];  // 64
  const float* b_vln   = (const float*)d_in[13];
  const float* w_q     = (const float*)d_in[14];  // 1024x1024
  const float* b_q     = (const float*)d_in[15];
  const float* w_o     = (const float*)d_in[16];  // 1024x1024
  const float* b_o     = (const float*)d_in[17];
  const float* w_f1    = (const float*)d_in[18];  // 4096x1024
  const float* b_f1    = (const float*)d_in[19];
  const float* w_f2    = (const float*)d_in[20];  // 1024x4096
  const float* b_f2    = (const float*)d_in[21];
  const float* g_ffln  = (const float*)d_in[22];
  const float* b_ffln  = (const float*)d_in[23];
  const float* w_fin   = (const float*)d_in[24];  // 128x1024
  const float* b_fin   = (const float*)d_in[25];

  float* ws = (float*)d_ws;
  float* DX = ws;                    // 544*128
  float* X  = DX + 544 * 128;        // 544*1024
  float* Kb = X  + 544 * 1024;       // 544*1024
  float* Vb = Kb + 544 * 1024;       // 544*1024
  float* Qb = Vb + 544 * 1024;       // 32*1024
  float* z  = Qb + 32 * 1024;        // 32*1024
  float* zo = z  + 32 * 1024;        // 32*1024
  float* zl = zo + 32 * 1024;        // 32*1024
  float* h1 = zl + 32 * 1024;        // 32*4096
  float* z2 = h1 + 32 * 4096;        // 32*1024   (total ~8.1 MB)

  const dim3 blk(256);

  // 1) initial_proj -> DX rows 0..31      (32x128 = 32x64 @ 64x128)
  gemm_bf16<0, 0><<<dim3(128 / 128, 32 / 32), blk, 0, stream>>>(
      initial, 64, w_init, b_init, nullptr, DX, 128, 64);
  // 2) logsig -> DX rows 32..543
  pack_logsig<<<dim3((32 * 16 * 128 + 255) / 256), blk, 0, stream>>>(logsig, DX);
  // 3) X = DX @ w_in^T + b_in             (544x1024)
  gemm_bf16<0, 0><<<dim3(1024 / 128, 544 / 32), blk, 0, stream>>>(
      DX, 128, w_in, b_in, nullptr, X, 1024, 128);
  // 4) X = LN(X) (in-place, per 1024-row)
  ln_rows_1024<<<dim3(544), blk, 0, stream>>>(X, X, g_inln, b_inln);
  // 5) Kpre / Vpre                        (544x1024 each, K=1024)
  gemm_bf16<0, 0><<<dim3(1024 / 128, 544 / 32), blk, 0, stream>>>(
      X, 1024, w_k, b_k, nullptr, Kb, 1024, 1024);
  gemm_bf16<0, 0><<<dim3(1024 / 128, 544 / 32), blk, 0, stream>>>(
      X, 1024, w_v, b_v, nullptr, Vb, 1024, 1024);
  // 6) Qpre from X slice 16 (= logsig[:,T-1] rows)
  gemm_bf16<0, 0><<<dim3(1024 / 128, 32 / 32), blk, 0, stream>>>(
      X + (size_t)512 * 1024, 1024, w_q, b_q, nullptr, Qb, 1024, 1024);
  // 7) per-head softmax / LN / softmax
  softmax_head64<<<dim3(544 * 16), dim3(64), 0, stream>>>(Kb);
  ln_head64<<<dim3(544 * 16), dim3(64), 0, stream>>>(Vb, g_vln, b_vln);
  softmax_head64<<<dim3(32 * 16), dim3(64), 0, stream>>>(Qb);
  // 8) RK4 Oja scan + readout z           (512 blocks of 64 threads)
  oja_scan<<<dim3(32 * 16), dim3(64), 0, stream>>>(Kb, Vb, Qb, z);
  // 9) zo = z @ w_o^T + b_o
  gemm_bf16<0, 0><<<dim3(1024 / 128, 32 / 32), blk, 0, stream>>>(
      z, 1024, w_o, b_o, nullptr, zo, 1024, 1024);
  // 10) zl = ff_ln(zo)
  ln_rows_1024<<<dim3(32), blk, 0, stream>>>(zo, zl, g_ffln, b_ffln);
  // 11) h1 = relu(zl @ w_f1^T + b_f1)     (32x4096)
  gemm_bf16<1, 0><<<dim3(4096 / 128, 32 / 32), blk, 0, stream>>>(
      zl, 1024, w_f1, b_f1, nullptr, h1, 4096, 1024);
  // 12) z2 = h1 @ w_f2^T + b_f2 + zo      (residual fused)
  gemm_bf16<0, 1><<<dim3(1024 / 128, 32 / 32), blk, 0, stream>>>(
      h1, 4096, w_f2, b_f2, zo, z2, 1024, 4096);
  // 13) out = z2 @ w_fin^T + b_fin        (32x128, fp32 output)
  gemm_bf16<0, 0><<<dim3(128 / 128, 32 / 32), blk, 0, stream>>>(
      z2, 1024, w_fin, b_fin, nullptr, (float*)d_out, 128, 1024);
}
